// MambaClass_63213328662777
// MI455X (gfx1250) — matrix operand order repaired and checked by reading, not compiled, not run on hardware
//
#include <hip/hip_runtime.h>
#include <hip/hip_bf16.h>
#include <math.h>

// ---------------------------------------------------------------------------
// Types for CDNA5 WMMA (gfx1250, wave32)
// ---------------------------------------------------------------------------
typedef unsigned short ushort_t;
typedef __attribute__((ext_vector_type(16))) __bf16 v16bf;
typedef __attribute__((ext_vector_type(8)))  float  v8f;

#define D_MODEL   388
#define N_BLOCKS  5
#define D_INNER   776
#define D_CONV    4
#define DT_RANK   25
#define D_STATE   16
#define SEQ       145
#define NUM_CLS   3
#define BATCH     256
#define M_ROWS    (BATCH * SEQ)          // 37120 = 128 * 290

// padded dims: K multiple of 32, N multiple of 64 (32x64 wave macro-tile)
#define KP_DM     416                    // 388  -> 416
#define KP_DI     800                    // 776  -> 800
#define KP_DT     32                     // 25   -> 32
#define NP_E      1600                   // 1552 -> 1600 (in_proj N, xz ld)
#define NP_XP     64                     // 57   -> 64
#define NP_DT     832                    // 776  -> 832
#define NP_OUT    448                    // 388  -> 448

__device__ __forceinline__ ushort_t f2bf(float f) {
  unsigned u = __builtin_bit_cast(unsigned, f);
  u += 0x7FFFu + ((u >> 16) & 1u);       // round-to-nearest-even
  return (ushort_t)(u >> 16);
}
__device__ __forceinline__ float silu(float x) {
  return x / (1.0f + __expf(-x));
}

// ---------------------------------------------------------------------------
// Register-blocked bf16 WMMA GEMM: C[M x N](f32) = A[M x Kp] * W[N x Kp]^T
// One wave32 computes a 32x64 macro-tile: 2 A-frags x 4 B-frags = 8 WMMAs
// per 32-wide K step. Block = 4 waves stacked along M (128 rows).
// Fragment layouts per CDNA5 ISA 7.12.2; all loads are b128.
// ---------------------------------------------------------------------------
__global__ __launch_bounds__(128)
void gemm_bf16_wmma(const ushort_t* __restrict__ A, int lda,
                    const ushort_t* __restrict__ B, int ldb,
                    float* __restrict__ C, int ldc, int K)
{
  const int wave = threadIdx.x >> 5;
  const int lane = threadIdx.x & 31;
  const int r    = lane & 15;
  const int hi   = lane >> 4;

  const int m0 = (blockIdx.y * 4 + wave) * 32;   // 32 rows per wave
  const int n0 = blockIdx.x * 64;                // 64 cols per wave

  // A 16x32 bf16 frag: lanes 0-15 row-major, K halves {hi*8.., 16+hi*8..}
  const ushort_t* Ap0 = A + (size_t)(m0 + r) * lda + hi * 8;
  const ushort_t* Ap1 = Ap0 + (size_t)16 * lda;
  // B 32x16 bf16 frag from row-major weight (N x Kp): lane = n, packed K pairs
  const ushort_t* Bp0 = B + (size_t)(n0 +  0 + r) * ldb + hi * 8;
  const ushort_t* Bp1 = B + (size_t)(n0 + 16 + r) * ldb + hi * 8;
  const ushort_t* Bp2 = B + (size_t)(n0 + 32 + r) * ldb + hi * 8;
  const ushort_t* Bp3 = B + (size_t)(n0 + 48 + r) * ldb + hi * 8;

  v8f acc[2][4] = {};
  union frag { uint4 q[2]; v16bf v; };

  for (int k = 0; k < K; k += 32) {
    frag a0, a1, b0, b1, b2, b3;
    a0.q[0] = *(const uint4*)(Ap0 + k);
    a0.q[1] = *(const uint4*)(Ap0 + k + 16);
    a1.q[0] = *(const uint4*)(Ap1 + k);
    a1.q[1] = *(const uint4*)(Ap1 + k + 16);
    b0.q[0] = *(const uint4*)(Bp0 + k);
    b0.q[1] = *(const uint4*)(Bp0 + k + 16);
    b1.q[0] = *(const uint4*)(Bp1 + k);
    b1.q[1] = *(const uint4*)(Bp1 + k + 16);
    b2.q[0] = *(const uint4*)(Bp2 + k);
    b2.q[1] = *(const uint4*)(Bp2 + k + 16);
    b3.q[0] = *(const uint4*)(Bp3 + k);
    b3.q[1] = *(const uint4*)(Bp3 + k + 16);

    acc[0][0] = __builtin_amdgcn_wmma_f32_16x16x32_bf16(false, a0.v, false, b0.v, (short)0, acc[0][0], false, false);
    acc[1][0] = __builtin_amdgcn_wmma_f32_16x16x32_bf16(false, a1.v, false, b0.v, (short)0, acc[1][0], false, false);
    acc[0][1] = __builtin_amdgcn_wmma_f32_16x16x32_bf16(false, a0.v, false, b1.v, (short)0, acc[0][1], false, false);
    acc[1][1] = __builtin_amdgcn_wmma_f32_16x16x32_bf16(false, a1.v, false, b1.v, (short)0, acc[1][1], false, false);
    acc[0][2] = __builtin_amdgcn_wmma_f32_16x16x32_bf16(false, a0.v, false, b2.v, (short)0, acc[0][2], false, false);
    acc[1][2] = __builtin_amdgcn_wmma_f32_16x16x32_bf16(false, a1.v, false, b2.v, (short)0, acc[1][2], false, false);
    acc[0][3] = __builtin_amdgcn_wmma_f32_16x16x32_bf16(false, a0.v, false, b3.v, (short)0, acc[0][3], false, false);
    acc[1][3] = __builtin_amdgcn_wmma_f32_16x16x32_bf16(false, a1.v, false, b3.v, (short)0, acc[1][3], false, false);
  }

#pragma unroll
  for (int i = 0; i < 2; ++i) {
    float* Cr = C + (size_t)(m0 + 16 * i + hi * 8) * ldc + n0 + r;
#pragma unroll
    for (int j = 0; j < 4; ++j)
#pragma unroll
      for (int rr = 0; rr < 8; ++rr)       // VGPR rr -> M = hi*8+rr
        Cr[(size_t)rr * ldc + 16 * j] = acc[i][j][rr];
  }
}

// ---------------------------------------------------------------------------
// Weight packing: f32 (N x K) -> bf16 (Npad x Kpad), zero pad.
// ---------------------------------------------------------------------------
__global__ __launch_bounds__(256)
void pack_w_bf16(const float* __restrict__ src, ushort_t* __restrict__ dst,
                 int N, int K, int Npad, int Kpad)
{
  long long idx = (long long)blockIdx.x * 256 + threadIdx.x;
  long long total = (long long)Npad * Kpad;
  if (idx >= total) return;
  int r = (int)(idx / Kpad), k = (int)(idx % Kpad);
  float v = (r < N && k < K) ? src[(size_t)r * K + k] : 0.0f;
  dst[idx] = f2bf(v);
}

__global__ __launch_bounds__(256)
void zero_pad_cols(ushort_t* __restrict__ P, int ld, int start, long long rows)
{
  long long idx = (long long)blockIdx.x * 256 + threadIdx.x;
  int pad = ld - start;
  long long total = rows * pad;
  if (idx >= total) return;
  long long row = idx / pad; int j = (int)(idx % pad);
  P[row * ld + start + j] = 0;
}

// ---------------------------------------------------------------------------
// LayerNorm over D_MODEL, output bf16 with zero-padded K tail.
// ---------------------------------------------------------------------------
__global__ __launch_bounds__(128)
void layernorm_to_bf16(const float* __restrict__ X, int ldx,
                       const float* __restrict__ g, const float* __restrict__ b,
                       ushort_t* __restrict__ H, int ldh, int D)
{
  __shared__ float red[128];
  const size_t row = blockIdx.x;
  const float* xr = X + row * (size_t)ldx;

  float s = 0.f;
  for (int j = threadIdx.x; j < D; j += 128) s += xr[j];
  red[threadIdx.x] = s; __syncthreads();
  for (int o = 64; o > 0; o >>= 1) {
    if (threadIdx.x < o) red[threadIdx.x] += red[threadIdx.x + o];
    __syncthreads();
  }
  const float mu = red[0] / D;
  __syncthreads();

  float v = 0.f;
  for (int j = threadIdx.x; j < D; j += 128) { float t = xr[j] - mu; v += t * t; }
  red[threadIdx.x] = v; __syncthreads();
  for (int o = 64; o > 0; o >>= 1) {
    if (threadIdx.x < o) red[threadIdx.x] += red[threadIdx.x + o];
    __syncthreads();
  }
  const float rstd = rsqrtf(red[0] / D + 1e-5f);

  ushort_t* hr = H + row * (size_t)ldh;
  for (int j = threadIdx.x; j < ldh; j += 128) {
    float o = (j < D) ? ((xr[j] - mu) * rstd * g[j] + b[j]) : 0.f;
    hr[j] = f2bf(o);
  }
}

// ---------------------------------------------------------------------------
// Causal depthwise conv(4) + bias + SiLU. xz rows of NP_E (cols 0..775 = xc,
// 776..1551 = z). Writes xc as f32 (scan input) and bf16 (x_proj input).
// ---------------------------------------------------------------------------
__global__ __launch_bounds__(256)
void conv_silu(const float* __restrict__ XZ, const float* __restrict__ cw,
               const float* __restrict__ cb, float* __restrict__ XCF,
               ushort_t* __restrict__ XCB, long long total)
{
  long long idx = (long long)blockIdx.x * 256 + threadIdx.x;
  if (idx >= total) return;
  int d = (int)(idx % D_INNER);
  long long row = idx / D_INNER;
  int l = (int)(row % SEQ);

  float acc = cb[d];
#pragma unroll
  for (int j = 0; j < D_CONV; ++j) {
    int lj = l - (D_CONV - 1) + j;
    if (lj >= 0)
      acc += XZ[(row - (D_CONV - 1) + j) * NP_E + d] * cw[d * D_CONV + j];
  }
  float s = silu(acc);
  XCF[row * D_INNER + d] = s;
  XCB[row * KP_DI + d]   = f2bf(s);
}

// dt slice (first 25 cols of dbl, ld NP_XP) -> bf16 (M x 32), zero padded
__global__ __launch_bounds__(256)
void dt_prep(const float* __restrict__ DBL, ushort_t* __restrict__ DTIN,
             long long total)
{
  long long idx = (long long)blockIdx.x * 256 + threadIdx.x;
  if (idx >= total) return;
  long long row = idx / KP_DT; int j = (int)(idx % KP_DT);
  DTIN[idx] = (j < DT_RANK) ? f2bf(DBL[row * NP_XP + j]) : (ushort_t)0;
}

// softplus(dt_raw + dt_b)
__global__ __launch_bounds__(256)
void dt_act(const float* __restrict__ RAW, const float* __restrict__ db,
            float* __restrict__ DT, long long total)
{
  long long idx = (long long)blockIdx.x * 256 + threadIdx.x;
  if (idx >= total) return;
  long long row = idx / D_INNER; int d = (int)(idx % D_INNER);
  float v = RAW[row * NP_DT + d] + db[d];
  DT[idx] = (v > 20.f) ? v : log1pf(__expf(v));
}

// ---------------------------------------------------------------------------
// Selective scan: one thread per (b,d) channel, 16-state in registers.
// Fuses +u*D, *silu(z), and bf16 convert for out_proj.
// ---------------------------------------------------------------------------
__global__ __launch_bounds__(256)
void selective_scan(const float* __restrict__ DT, const float* __restrict__ U,
                    const float* __restrict__ DBL, const float* __restrict__ XZ,
                    const float* __restrict__ A_log, const float* __restrict__ Dp,
                    ushort_t* __restrict__ YBF)
{
  int d = blockIdx.x * 256 + threadIdx.x;
  if (d >= D_INNER) return;
  int b = blockIdx.y;

  float An[D_STATE];
#pragma unroll
  for (int n = 0; n < D_STATE; ++n) An[n] = -__expf(A_log[(size_t)d * D_STATE + n]);
  const float Dv = Dp[d];

  float h[D_STATE];
#pragma unroll
  for (int n = 0; n < D_STATE; ++n) h[n] = 0.f;

  for (int l = 0; l < SEQ; ++l) {
    size_t row = (size_t)b * SEQ + l;
    float dt = DT[row * D_INNER + d];
    float u  = U[row * D_INNER + d];
    float du = dt * u;
    const float* bc = DBL + row * NP_XP;
    float y = 0.f;
#pragma unroll
    for (int n = 0; n < D_STATE; ++n) {
      h[n] = __expf(dt * An[n]) * h[n] + du * bc[DT_RANK + n];
      y   += h[n] * bc[DT_RANK + D_STATE + n];
    }
    y += u * Dv;
    float z = XZ[row * NP_E + D_INNER + d];
    y *= silu(z);
    YBF[row * KP_DI + d] = f2bf(y);
  }
}

// ---------------------------------------------------------------------------
// Classifier: logits = flat(x) @ cls_w^T + b, softmax over 3.
// ---------------------------------------------------------------------------
__global__ __launch_bounds__(256)
void classifier_softmax(const float* __restrict__ X,        // ld NP_OUT
                        const float* __restrict__ W,        // (3, SEQ*D_MODEL)
                        const float* __restrict__ bias,
                        float* __restrict__ out)
{
  __shared__ float red[NUM_CLS][256];
  const int b = blockIdx.x;
  const int F = SEQ * D_MODEL;           // 56260
  float s0 = 0.f, s1 = 0.f, s2 = 0.f;
  for (int idx = threadIdx.x; idx < F; idx += 256) {
    int l = idx / D_MODEL, dd = idx % D_MODEL;
    float v = X[((size_t)b * SEQ + l) * NP_OUT + dd];
    s0 += v * W[idx];
    s1 += v * W[F + idx];
    s2 += v * W[2 * F + idx];
  }
  red[0][threadIdx.x] = s0; red[1][threadIdx.x] = s1; red[2][threadIdx.x] = s2;
  __syncthreads();
  for (int o = 128; o > 0; o >>= 1) {
    if (threadIdx.x < o)
      for (int c = 0; c < NUM_CLS; ++c)
        red[c][threadIdx.x] += red[c][threadIdx.x + o];
    __syncthreads();
  }
  if (threadIdx.x == 0) {
    float lg[NUM_CLS];
    float mx = -1e30f;
    for (int c = 0; c < NUM_CLS; ++c) { lg[c] = red[c][0] + bias[c]; mx = fmaxf(mx, lg[c]); }
    float sum = 0.f;
    for (int c = 0; c < NUM_CLS; ++c) { lg[c] = __expf(lg[c] - mx); sum += lg[c]; }
    for (int c = 0; c < NUM_CLS; ++c) out[(size_t)b * NUM_CLS + c] = lg[c] / sum;
  }
}

// ---------------------------------------------------------------------------
// Host-side orchestration
// ---------------------------------------------------------------------------
static inline int cdiv_ll(long long a, long long b) { return (int)((a + b - 1) / b); }

extern "C" void kernel_launch(void* const* d_in, const int* in_sizes, int n_in,
                              void* d_out, int out_size, void* d_ws, size_t ws_size,
                              hipStream_t stream) {
  const float* x_in   = (const float*)d_in[0];
  const float* ln_g   = (const float*)d_in[1];
  const float* ln_b   = (const float*)d_in[2];
  const float* w_in   = (const float*)d_in[3];
  const float* conv_w = (const float*)d_in[4];
  const float* conv_b = (const float*)d_in[5];
  const float* w_xp   = (const float*)d_in[6];
  const float* w_dt   = (const float*)d_in[7];
  const float* dt_b   = (const float*)d_in[8];
  const float* A_log  = (const float*)d_in[9];
  const float* Dvec   = (const float*)d_in[10];
  const float* w_out  = (const float*)d_in[11];
  const float* cls_w  = (const float*)d_in[12];
  const float* cls_b  = (const float*)d_in[13];

  const long long M = M_ROWS;
  char* p = (char*)d_ws;
  auto take = [&](size_t bytes) { char* q = p; p += (bytes + 255) & ~(size_t)255; return q; };
  ushort_t* hbf   = (ushort_t*)take((size_t)M * KP_DM * 2);
  float*    xz    = (float*)   take((size_t)M * NP_E * 4);
  float*    xcf   = (float*)   take((size_t)M * D_INNER * 4);
  ushort_t* xcb   = (ushort_t*)take((size_t)M * KP_DI * 2);
  float*    dbl   = (float*)   take((size_t)M * NP_XP * 4);
  ushort_t* dtin  = (ushort_t*)take((size_t)M * KP_DT * 2);
  float*    dtraw = (float*)   take((size_t)M * NP_DT * 4);
  float*    dtv   = (float*)   take((size_t)M * D_INNER * 4);
  ushort_t* ybf   = (ushort_t*)take((size_t)M * KP_DI * 2);
  float*    xb0   = (float*)   take((size_t)M * NP_OUT * 4);
  float*    xb1   = (float*)   take((size_t)M * NP_OUT * 4);
  ushort_t* wb_in = (ushort_t*)take((size_t)N_BLOCKS * NP_E * KP_DM * 2);
  ushort_t* wb_xp = (ushort_t*)take((size_t)N_BLOCKS * NP_XP * KP_DI * 2);
  ushort_t* wb_dt = (ushort_t*)take((size_t)N_BLOCKS * NP_DT * KP_DT * 2);
  ushort_t* wb_out= (ushort_t*)take((size_t)N_BLOCKS * NP_OUT * KP_DI * 2);

  // pack all weights to padded bf16 (cheap, per call -> deterministic)
  for (int i = 0; i < N_BLOCKS; ++i) {
    pack_w_bf16<<<cdiv_ll((long long)NP_E * KP_DM, 256), 256, 0, stream>>>(
        w_in + (size_t)i * 2 * D_INNER * D_MODEL, wb_in + (size_t)i * NP_E * KP_DM,
        2 * D_INNER, D_MODEL, NP_E, KP_DM);
    pack_w_bf16<<<cdiv_ll((long long)NP_XP * KP_DI, 256), 256, 0, stream>>>(
        w_xp + (size_t)i * (DT_RANK + 2 * D_STATE) * D_INNER,
        wb_xp + (size_t)i * NP_XP * KP_DI,
        DT_RANK + 2 * D_STATE, D_INNER, NP_XP, KP_DI);
    pack_w_bf16<<<cdiv_ll((long long)NP_DT * KP_DT, 256), 256, 0, stream>>>(
        w_dt + (size_t)i * D_INNER * DT_RANK, wb_dt + (size_t)i * NP_DT * KP_DT,
        D_INNER, DT_RANK, NP_DT, KP_DT);
    pack_w_bf16<<<cdiv_ll((long long)NP_OUT * KP_DI, 256), 256, 0, stream>>>(
        w_out + (size_t)i * D_MODEL * D_INNER, wb_out + (size_t)i * NP_OUT * KP_DI,
        D_MODEL, D_INNER, NP_OUT, KP_DI);
  }

  const int TB = (int)(M / 128);         // 290 macro-row blocks (4 waves x 32)
  const float* xcur = x_in;
  int ldx = D_MODEL;
  float* xbuf[2] = { xb0, xb1 };

  for (int i = 0; i < N_BLOCKS; ++i) {
    // 1) LayerNorm -> bf16 (padded 416)
    layernorm_to_bf16<<<(int)M, 128, 0, stream>>>(
        xcur, ldx, ln_g + (size_t)i * D_MODEL, ln_b + (size_t)i * D_MODEL,
        hbf, KP_DM, D_MODEL);

    // 2) in_proj: (M x 416) x (1600 x 416)^T -> xz (M x 1600)   [WMMA]
    gemm_bf16_wmma<<<dim3(NP_E / 64, TB), 128, 0, stream>>>(
        hbf, KP_DM, wb_in + (size_t)i * NP_E * KP_DM, KP_DM, xz, NP_E, KP_DM);

    // 3) causal depthwise conv + SiLU
    conv_silu<<<cdiv_ll(M * D_INNER, 256), 256, 0, stream>>>(
        xz, conv_w + (size_t)i * D_INNER * D_CONV, conv_b + (size_t)i * D_INNER,
        xcf, xcb, M * D_INNER);
    zero_pad_cols<<<cdiv_ll(M * (KP_DI - D_INNER), 256), 256, 0, stream>>>(
        xcb, KP_DI, D_INNER, M);

    // 4) x_proj: (M x 800) x (64 x 800)^T -> dbl (M x 64)       [WMMA]
    gemm_bf16_wmma<<<dim3(NP_XP / 64, TB), 128, 0, stream>>>(
        xcb, KP_DI, wb_xp + (size_t)i * NP_XP * KP_DI, KP_DI, dbl, NP_XP, KP_DI);

    // 5) dt_proj: (M x 32) x (832 x 32)^T -> dtraw (M x 832)    [WMMA]
    dt_prep<<<cdiv_ll(M * KP_DT, 256), 256, 0, stream>>>(dbl, dtin, M * KP_DT);
    gemm_bf16_wmma<<<dim3(NP_DT / 64, TB), 128, 0, stream>>>(
        dtin, KP_DT, wb_dt + (size_t)i * NP_DT * KP_DT, KP_DT, dtraw, NP_DT, KP_DT);
    dt_act<<<cdiv_ll(M * D_INNER, 256), 256, 0, stream>>>(
        dtraw, dt_b + (size_t)i * D_INNER, dtv, M * D_INNER);

    // 6) selective scan (fused D-skip, SiLU gate, bf16 convert)
    zero_pad_cols<<<cdiv_ll(M * (KP_DI - D_INNER), 256), 256, 0, stream>>>(
        ybf, KP_DI, D_INNER, M);
    selective_scan<<<dim3(cdiv_ll(D_INNER, 256), BATCH), 256, 0, stream>>>(
        dtv, xcf, dbl, xz, A_log + (size_t)i * D_INNER * D_STATE,
        Dvec + (size_t)i * D_INNER, ybf);

    // 7) out_proj: (M x 800) x (448 x 800)^T -> x_next (M x 448) [WMMA]
    float* xn = xbuf[i & 1];
    gemm_bf16_wmma<<<dim3(NP_OUT / 64, TB), 128, 0, stream>>>(
        ybf, KP_DI, wb_out + (size_t)i * NP_OUT * KP_DI, KP_DI, xn, NP_OUT, KP_DI);
    xcur = xn;
    ldx  = NP_OUT;
  }

  // 8) classifier + softmax
  classifier_softmax<<<BATCH, 256, 0, stream>>>(xcur, cls_w, cls_b, (float*)d_out);
}
